// mLSTMLayer_81741817577595
// MI455X (gfx1250) — matrix operand order repaired and checked by reading, not compiled, not run on hardware
//
#include <hip/hip_runtime.h>
#include <hip/hip_bf16.h>
#include <math.h>

// ---------------------------------------------------------------------------
// mLSTM layer for gfx1250 (MI455X).
// Matrix products: v_wmma_f32_16x16x32_bf16 (fp32 accumulate).
// GEMM staging: double-buffered ASYNC global->LDS copies
// (global_load_async_to_lds_b128 / s_wait_asynccnt, ISA ch.10.7 & 15.18).
// ---------------------------------------------------------------------------

typedef __attribute__((ext_vector_type(16))) __bf16 v16bf;
typedef __attribute__((ext_vector_type(8)))  __bf16 v8bf;
typedef __attribute__((ext_vector_type(4)))  __bf16 v4bf;
typedef __attribute__((ext_vector_type(8)))  float  v8f;

constexpr int Bc     = 2;
constexpr int Sc     = 2048;
constexpr int Ec     = 1024;
constexpr int INNERc = 2048;
constexpr int NHc    = 4;
constexpr int DHc    = 512;   // INNER / NH
constexpr int Kc     = 4;     // conv taps
constexpr float EPSc = 1e-6f;

#define SHUF16(lo, hi) __builtin_shufflevector(lo, hi, 0, 1, 2, 3, 4, 5, 6, 7, \
                                               8, 9, 10, 11, 12, 13, 14, 15)

// ---- CDNA5 async global->LDS copy (per-lane 16B), tracked by ASYNCcnt ------
__device__ __forceinline__ void async_copy_b128(const void* g, void* lds) {
  // VDST = per-lane LDS byte address (low 32 bits of generic pointer),
  // VADDR = 64-bit global address. GV addressing mode, SADDR=NULL.
  asm volatile("global_load_async_to_lds_b128 %0, %1, off"
               :
               : "v"((unsigned)(size_t)lds),
                 "v"((unsigned long long)(size_t)g)
               : "memory");
}
__device__ __forceinline__ void wait_async0() {
  asm volatile("s_wait_asynccnt 0x0" ::: "memory");
}

// ===========================================================================
// fp32 -> bf16 (4 elements / thread)
// ===========================================================================
__global__ __launch_bounds__(256) void cvt_bf16_kernel(
    const float* __restrict__ in, __bf16* __restrict__ out) {
  const size_t i = ((size_t)blockIdx.x * 256 + threadIdx.x) * 4;
  const float4 f = *(const float4*)(in + i);
  v4bf o;
  o[0] = (__bf16)f.x; o[1] = (__bf16)f.y;
  o[2] = (__bf16)f.z; o[3] = (__bf16)f.w;
  *(v4bf*)(out + i) = o;
}

// ===========================================================================
// W (Krows x Ncols, fp32) -> Wt (Ncols x Krows, bf16), LDS tiled transpose
// ===========================================================================
__global__ __launch_bounds__(256) void transpose_cvt_kernel(
    const float* __restrict__ W, __bf16* __restrict__ Wt, int Krows,
    int Ncols) {
  __shared__ float tile[32][33];
  const int k0 = blockIdx.y * 32, n0 = blockIdx.x * 32;
  const int tx = threadIdx.x & 31, ty = threadIdx.x >> 5;  // 32 x 8
#pragma unroll
  for (int i = 0; i < 32; i += 8)
    tile[ty + i][tx] = W[(size_t)(k0 + ty + i) * Ncols + n0 + tx];
  __syncthreads();
#pragma unroll
  for (int i = 0; i < 32; i += 8)
    Wt[(size_t)(n0 + ty + i) * Krows + k0 + tx] = (__bf16)tile[tx][ty + i];
}

// ===========================================================================
// GEMM: C[M,N] = A[M,K] x Bt[N,K]^T, A/Bt bf16 row-major, C fp32.
// 256 threads = 8 waves; block tile 128x128, K-step 32; wave tile 32x64.
// K-slabs fetched with double-buffered async global->LDS copies; fragments
// read back as aligned ds_load_b128 in contiguous k order (16*half + s), same for A and B.
// ===========================================================================
__global__ __launch_bounds__(256) void gemm_bf16_wmma(
    const __bf16* __restrict__ A, const __bf16* __restrict__ Bt,
    float* __restrict__ C, int M, int N, int K) {
  __shared__ __align__(64) __bf16 Asl[2][128][32];
  __shared__ __align__(64) __bf16 Bsl[2][128][32];

  const int wave = threadIdx.x >> 5;
  const int lane = threadIdx.x & 31;
  const int half = lane >> 4;
  const int l15  = lane & 15;
  const int mw   = wave >> 1;          // 0..3 -> 32-row slice
  const int nw   = wave & 1;           // 0..1 -> 64-col slice
  const int m0   = blockIdx.y * 128;
  const int n0   = blockIdx.x * 128;
  const int rm   = mw * 32;
  const int cn   = nw * 64;
  const int trow = threadIdx.x >> 1;   // staging: 128 rows x 2 segments
  const int tseg = threadIdx.x & 1;

  const __bf16* aRow = A + (size_t)(m0 + trow) * K + tseg * 16;
  const __bf16* bRow = Bt + (size_t)(n0 + trow) * K + tseg * 16;

  auto stage = [&](int buf, int k0) {
    async_copy_b128(aRow + k0,     &Asl[buf][trow][tseg * 16]);
    async_copy_b128(aRow + k0 + 8, &Asl[buf][trow][tseg * 16 + 8]);
    async_copy_b128(bRow + k0,     &Bsl[buf][trow][tseg * 16]);
    async_copy_b128(bRow + k0 + 8, &Bsl[buf][trow][tseg * 16 + 8]);
  };

  v8f zero = {};
  v8f acc[2][4] = {{zero, zero, zero, zero}, {zero, zero, zero, zero}};

  stage(0, 0);                         // prologue fill of buffer 0
  int buf = 0;
  for (int k0 = 0; k0 < K; k0 += 32, buf ^= 1) {
    wait_async0();                     // this wave's slab has landed in LDS
    __syncthreads();                   // ... and so has everyone else's
    if (k0 + 32 < K) stage(buf ^ 1, k0 + 32);  // overlap next fetch
    if (k0 + 64 < K) {                 // L2 prefetch for the slab after next
      __builtin_prefetch(aRow + k0 + 64, 0, 0);
      __builtin_prefetch(bRow + k0 + 64, 0, 0);
    }

    // ---- fragments from LDS
    v16bf afr[2];
#pragma unroll
    for (int mi = 0; mi < 2; ++mi) {
      const __bf16* ar = &Asl[buf][rm + mi * 16 + l15][0];
      const v8bf lo = *(const v8bf*)(ar + half * 16);
      const v8bf hi = *(const v8bf*)(ar + half * 16 + 8);
      afr[mi] = SHUF16(lo, hi);
    }
    v16bf bfrg[4];
#pragma unroll
    for (int ni = 0; ni < 4; ++ni)
      bfrg[ni] = *(const v16bf*)(&Bsl[buf][cn + ni * 16 + l15][half * 16]);

#pragma unroll
    for (int mi = 0; mi < 2; ++mi)
#pragma unroll
      for (int ni = 0; ni < 4; ++ni)
        acc[mi][ni] = __builtin_amdgcn_wmma_f32_16x16x32_bf16(
            false, afr[mi], false, bfrg[ni], (short)0, acc[mi][ni], false,
            false);
  }

  // ---- store C (row m = r + 8*half, col n = l15)
#pragma unroll
  for (int mi = 0; mi < 2; ++mi)
#pragma unroll
    for (int ni = 0; ni < 4; ++ni)
#pragma unroll
      for (int r = 0; r < 8; ++r)
        C[(size_t)(m0 + rm + mi * 16 + r + 8 * half) * N + n0 + cn + ni * 16 +
          l15] = acc[mi][ni][r];
}

// ===========================================================================
// Causal depthwise conv (K=4) + swish
// ===========================================================================
__global__ __launch_bounds__(256) void conv_swish_kernel(
    const float* __restrict__ xm, const float* __restrict__ kern,
    const float* __restrict__ bias, float* __restrict__ act_out) {
  const size_t idx = (size_t)blockIdx.x * 256 + threadIdx.x;
  const size_t row = idx / INNERc;
  const int c = (int)(idx % INNERc);
  const int s = (int)(row % Sc);
  float acc = bias[c];
#pragma unroll
  for (int t = 0; t < Kc; ++t) {
    const int sp = s - (Kc - 1) + t;
    if (sp >= 0)
      acc += kern[t * INNERc + c] * xm[(row - (Kc - 1) + t) * INNERc + c];
  }
  const float sg = 1.f / (1.f + __expf(-acc));
  act_out[idx] = acc * sg;
}

// ===========================================================================
// Headwise 4x4 projections -> bf16 q,k,v  (+ transposed v for the PV GEMM)
// ===========================================================================
__global__ __launch_bounds__(256) void headwise_kernel(
    const float* __restrict__ act, const float* __restrict__ xm,
    const float* __restrict__ Wq, const float* __restrict__ Wk,
    const float* __restrict__ Wv, __bf16* __restrict__ q,
    __bf16* __restrict__ k, __bf16* __restrict__ v,
    __bf16* __restrict__ vt) {
  const size_t idx = (size_t)blockIdx.x * 256 + threadIdx.x;
  const size_t row = idx / INNERc;
  const int c = (int)(idx % INNERc);
  const int b = (int)(row / Sc);
  const int s = (int)(row % Sc);
  const int n = c >> 2, dd = c & 3;
  const size_t rb = row * (size_t)INNERc + (size_t)(n << 2);
  const float* wq = Wq + n * 16 + dd;
  const float* wk = Wk + n * 16 + dd;
  const float* wv = Wv + n * 16 + dd;
  float aq = 0.f, ak = 0.f, av = 0.f;
#pragma unroll
  for (int kk = 0; kk < 4; ++kk) {
    const float a  = act[rb + kk];
    const float xv = xm[rb + kk];
    aq += a * wq[kk * 4];
    ak += a * wk[kk * 4];
    av += xv * wv[kk * 4];
  }
  q[idx] = (__bf16)aq;
  k[idx] = (__bf16)ak;
  v[idx] = (__bf16)av;
  vt[((size_t)b * INNERc + c) * Sc + s] = (__bf16)av;
}

// ===========================================================================
// Gate preactivations from bf16 q,k,v
// ===========================================================================
__global__ __launch_bounds__(256) void gates_kernel(
    const __bf16* __restrict__ q, const __bf16* __restrict__ k,
    const __bf16* __restrict__ v, const float* __restrict__ Wi,
    const float* __restrict__ bi, const float* __restrict__ Wf,
    const float* __restrict__ bf, float* __restrict__ ipre,
    float* __restrict__ fpre) {
  const int s = blockIdx.x, h = blockIdx.y, b = blockIdx.z;
  __shared__ float si[256], sf[256];
  const size_t base = ((size_t)b * Sc + s) * INNERc;
  float ai = 0.f, af = 0.f;
  for (int f = threadIdx.x; f < 3 * INNERc; f += 256) {
    float g;
    if (f < INNERc)            g = (float)q[base + f];
    else if (f < 2 * INNERc)   g = (float)k[base + f - INNERc];
    else                       g = (float)v[base + f - 2 * INNERc];
    ai += g * Wi[(size_t)f * NHc + h];
    af += g * Wf[(size_t)f * NHc + h];
  }
  si[threadIdx.x] = ai; sf[threadIdx.x] = af;
  __syncthreads();
  for (int st = 128; st > 0; st >>= 1) {
    if ((int)threadIdx.x < st) {
      si[threadIdx.x] += si[threadIdx.x + st];
      sf[threadIdx.x] += sf[threadIdx.x + st];
    }
    __syncthreads();
  }
  if (threadIdx.x == 0) {
    const size_t o = ((size_t)b * NHc + h) * Sc + s;
    ipre[o] = si[0] + bi[h];
    fpre[o] = sf[0] + bf[h];
  }
}

// ===========================================================================
// log_sigmoid + sequential cumsum per (b,h)
// ===========================================================================
__global__ void cumsum_kernel(const float* __restrict__ fpre,
                              float* __restrict__ lfc) {
  const int id = threadIdx.x;
  if (id >= Bc * NHc) return;
  float csum = 0.f;
  for (int s = 0; s < Sc; ++s) {
    const float xv = fpre[(size_t)id * Sc + s];
    const float ls = fminf(xv, 0.f) - log1pf(__expf(-fabsf(xv)));
    csum += ls;
    lfc[(size_t)id * Sc + s] = csum;
  }
}

// ===========================================================================
// Flash-style stabilized mLSTM attention (bf16 operands, wide aligned loads).
// Block = (row tile 16, head, batch), 8 waves; wave owns a 64-wide DH slice.
// ===========================================================================
__global__ __launch_bounds__(256) void attn_kernel(
    const __bf16* __restrict__ qm, const __bf16* __restrict__ km,
    const __bf16* __restrict__ vt, const float* __restrict__ ipre,
    const float* __restrict__ lfcum, float* __restrict__ hbuf) {
  const int rt = blockIdx.x;
  const int hh = blockIdx.y;
  const int b  = blockIdx.z;
  const int wave = threadIdx.x >> 5;
  const int lane = threadIdx.x & 31;
  const int half = lane >> 4;
  const int l15  = lane & 15;
  const int r0 = rt * 16;
  const float qkscale = 0.04419417382f;  // 1/sqrt(DH)

  __shared__ __align__(32) __bf16 plds[8][16][32];

  const size_t headoff = (size_t)hh * DHc;

  // ---- preload Q fragments (16B aligned v8bf pairs)
  v16bf qf[16];
#pragma unroll
  for (int kk = 0; kk < 16; ++kk) {
    const __bf16* arow =
        qm + ((size_t)b * Sc + (r0 + l15)) * INNERc + headoff + kk * 32;
    const v8bf lo = *(const v8bf*)(arow + half * 16);
    const v8bf hi = *(const v8bf*)(arow + half * 16 + 8);
    qf[kk] = SHUF16(lo, hi);
  }

  v8f zero = {};
  v8f acc[4] = {zero, zero, zero, zero};
  float Mrun[8], ssum[8];
#pragma unroll
  for (int r = 0; r < 8; ++r) { Mrun[r] = -3.0e38f; ssum[r] = 0.f; }

  const float* lfbh = lfcum + ((size_t)b * NHc + hh) * Sc;
  const float* ipbh = ipre  + ((size_t)b * NHc + hh) * Sc;

  const int ntile = (r0 + 16 + 31) / 32;
  for (int t = 0; t < ntile; ++t) {
    const int c0 = t * 32;

    // ---- S tile = Q K^T : two 16x16 subtiles, K frag = one 32B load
    v8f sc[2];
#pragma unroll
    for (int sub = 0; sub < 2; ++sub) {
      v8f s = zero;
#pragma unroll
      for (int kk = 0; kk < 16; ++kk) {
        const v16bf bfr = *(const v16bf*)(
            km + ((size_t)b * Sc + (c0 + sub * 16 + l15)) * INNERc + headoff +
            kk * 32 + half * 16);
        s = __builtin_amdgcn_wmma_f32_16x16x32_bf16(
            false, qf[kk], false, bfr, (short)0, s, false, false);
      }
      sc[sub] = s;
    }

    // ---- decay matrix, online max / rescale, P = score*exp(D-M')
    float pval[2][8];
#pragma unroll
    for (int r = 0; r < 8; ++r) {
      const int i = r0 + r + 8 * half;
      const float lfi = lfbh[i];
      float dm = -3.0e38f;
      float dv[2];
#pragma unroll
      for (int sub = 0; sub < 2; ++sub) {
        const int j = c0 + sub * 16 + l15;
        dv[sub] = (j <= i) ? (lfi - lfbh[j] + ipbh[j]) : -3.0e38f;
        dm = fmaxf(dm, dv[sub]);
      }
#pragma unroll
      for (int off = 1; off < 16; off <<= 1)
        dm = fmaxf(dm, __shfl_xor(dm, off, 32));
      const float Mnew = fmaxf(Mrun[r], dm);
      const float scl = (Mrun[r] <= -1e37f) ? 0.f : __expf(Mrun[r] - Mnew);
      float rs = 0.f;
#pragma unroll
      for (int sub = 0; sub < 2; ++sub) {
        const float p = (dv[sub] <= -1e37f)
                            ? 0.f
                            : sc[sub][r] * qkscale * __expf(dv[sub] - Mnew);
        pval[sub][r] = p;
        rs += p;
      }
#pragma unroll
      for (int off = 1; off < 16; off <<= 1)
        rs += __shfl_xor(rs, off, 32);
      ssum[r] = ssum[r] * scl + rs;
#pragma unroll
      for (int c = 0; c < 4; ++c) acc[c][r] *= scl;
      Mrun[r] = Mnew;
    }

    // ---- stage P through LDS (C layout -> A layout)
    __syncthreads();
#pragma unroll
    for (int r = 0; r < 8; ++r) {
      const int m = r + 8 * half;
      plds[wave][m][l15]      = (__bf16)pval[0][r];
      plds[wave][m][16 + l15] = (__bf16)pval[1][r];
    }
    __syncthreads();

    const __bf16* pr = &plds[wave][l15][0];
    const v8bf plo = *(const v8bf*)(pr + half * 16);
    const v8bf phi = *(const v8bf*)(pr + half * 16 + 8);
    const v16bf af = SHUF16(plo, phi);

    // ---- P * V : V^T frag = one 32B load (seq-contiguous)
#pragma unroll
    for (int c = 0; c < 4; ++c) {
      const v16bf bfr = *(const v16bf*)(
          vt + ((size_t)b * INNERc + headoff + wave * 64 + c * 16 + l15) * Sc +
          c0 + half * 16);
      acc[c] = __builtin_amdgcn_wmma_f32_16x16x32_bf16(
          false, af, false, bfr, (short)0, acc[c], false, false);
    }
  }

  // ---- epilogue: h = acc / (max(|ssum|, exp(-M)) + eps)
#pragma unroll
  for (int r = 0; r < 8; ++r) {
    const float nv  = fmaxf(fabsf(ssum[r]), __expf(-Mrun[r]));
    const float inv = 1.f / (nv + EPSc);
    const int i = r0 + r + 8 * half;
    float* out = hbuf + ((size_t)b * Sc + i) * INNERc + headoff + wave * 64;
#pragma unroll
    for (int c = 0; c < 4; ++c) out[c * 16 + l15] = acc[c][r] * inv;
  }
}

// ===========================================================================
// Multi-head layernorm + learnable skip + swish(z) gate -> bf16 h_state
// ===========================================================================
__global__ __launch_bounds__(256) void ln_gate_kernel(
    const float* __restrict__ h, const float* __restrict__ act,
    const float* __restrict__ z, const float* __restrict__ nsc,
    const float* __restrict__ skp, __bf16* __restrict__ hstate) {
  const int s = blockIdx.x, head = blockIdx.y, b = blockIdx.z;
  const int t = threadIdx.x;
  __shared__ float red[256];
  const size_t rowbase = ((size_t)b * Sc + s) * INNERc;
  const size_t base = rowbase + (size_t)head * DHc;
  const float v0 = h[base + t], v1 = h[base + t + 256];

  red[t] = v0 + v1;
  __syncthreads();
  for (int st = 128; st > 0; st >>= 1) {
    if (t < st) red[t] += red[t + st];
    __syncthreads();
  }
  const float mu = red[0] / (float)DHc;
  __syncthreads();
  const float d0 = v0 - mu, d1 = v1 - mu;
  red[t] = d0 * d0 + d1 * d1;
  __syncthreads();
  for (int st = 128; st > 0; st >>= 1) {
    if (t < st) red[t] += red[t + st];
    __syncthreads();
  }
  const float inv = rsqrtf(red[0] / (float)DHc + EPSc);

#pragma unroll
  for (int e = 0; e < 2; ++e) {
    const int dd = t + e * 256;
    const int gc = head * DHc + dd;
    const float hv = e ? v1 : v0;
    const float hn = (hv - mu) * inv * nsc[gc];
    const float hs = hn + skp[gc] * act[rowbase + gc];
    const float zv = z[rowbase + gc];
    const float sw = zv / (1.f + __expf(-zv));
    hstate[rowbase + gc] = (__bf16)(hs * sw);
  }
}

// ===========================================================================
// launch
// ===========================================================================
extern "C" void kernel_launch(void* const* d_in, const int* in_sizes, int n_in,
                              void* d_out, int out_size, void* d_ws,
                              size_t ws_size, hipStream_t stream) {
  (void)in_sizes; (void)n_in; (void)out_size; (void)ws_size;
  const float* x       = (const float*)d_in[0];
  const float* W_up_m  = (const float*)d_in[1];
  const float* W_up_z  = (const float*)d_in[2];
  const float* convk   = (const float*)d_in[3];
  const float* convb   = (const float*)d_in[4];
  const float* Wq      = (const float*)d_in[5];
  const float* Wk      = (const float*)d_in[6];
  const float* Wv      = (const float*)d_in[7];
  const float* Wi      = (const float*)d_in[8];
  const float* bi      = (const float*)d_in[9];
  const float* Wf      = (const float*)d_in[10];
  const float* bf      = (const float*)d_in[11];
  const float* nsc     = (const float*)d_in[12];
  const float* skp     = (const float*)d_in[13];
  const float* Wd      = (const float*)d_in[14];
  float* out = (float*)d_out;

  // ---- workspace layout
  const size_t NElem = (size_t)Bc * Sc * INNERc;  // 8,388,608
  char* cur = (char*)d_ws;
  auto alloc = [&](size_t bytes) {
    char* p = cur;
    cur += (bytes + 255) & ~(size_t)255;
    return p;
  };
  float*  xm    = (float*)alloc(NElem * 4);                 // x_mlstm, later h
  float*  zb    = (float*)alloc(NElem * 4);                 // z
  float*  actb  = (float*)alloc(NElem * 4);                 // conv activation
  __bf16* qbf   = (__bf16*)alloc(NElem * 2);
  __bf16* kbf   = (__bf16*)alloc(NElem * 2);
  __bf16* vbf   = (__bf16*)alloc(NElem * 2);
  __bf16* vtbf  = (__bf16*)alloc(NElem * 2);                // V transposed
  __bf16* hstbf = (__bf16*)alloc(NElem * 2);                // h_state (bf16)
  __bf16* xbf   = (__bf16*)alloc((size_t)Bc * Sc * Ec * 2); // x (bf16)
  __bf16* Wtm   = (__bf16*)alloc((size_t)Ec * INNERc * 2);  // W_up_m^T
  __bf16* Wtz   = (__bf16*)alloc((size_t)Ec * INNERc * 2);  // W_up_z^T
  __bf16* Wdt   = (__bf16*)alloc((size_t)INNERc * Ec * 2);  // W_down^T
  float*  ipre  = (float*)alloc((size_t)Bc * NHc * Sc * 4);
  float*  fpre  = (float*)alloc((size_t)Bc * NHc * Sc * 4);
  float*  lfc   = (float*)alloc((size_t)Bc * NHc * Sc * 4);

  const int M = Bc * Sc;  // 4096

  // ---- precision/layout prepass
  cvt_bf16_kernel<<<(int)((size_t)M * Ec / 1024), 256, 0, stream>>>(x, xbf);
  transpose_cvt_kernel<<<dim3(INNERc / 32, Ec / 32), 256, 0, stream>>>(
      W_up_m, Wtm, Ec, INNERc);
  transpose_cvt_kernel<<<dim3(INNERc / 32, Ec / 32), 256, 0, stream>>>(
      W_up_z, Wtz, Ec, INNERc);
  transpose_cvt_kernel<<<dim3(Ec / 32, INNERc / 32), 256, 0, stream>>>(
      Wd, Wdt, INNERc, Ec);

  // ---- up projections (4096 x 1024) x (1024 x 2048)
  gemm_bf16_wmma<<<dim3(INNERc / 128, M / 128), 256, 0, stream>>>(
      xbf, Wtm, xm, M, INNERc, Ec);
  gemm_bf16_wmma<<<dim3(INNERc / 128, M / 128), 256, 0, stream>>>(
      xbf, Wtz, zb, M, INNERc, Ec);

  // ---- conv + swish, headwise q/k/v (+ V^T)
  {
    const int blocks = (int)(NElem / 256);  // 32768
    conv_swish_kernel<<<blocks, 256, 0, stream>>>(xm, convk, convb, actb);
    headwise_kernel<<<blocks, 256, 0, stream>>>(actb, xm, Wq, Wk, Wv, qbf,
                                                kbf, vbf, vtbf);
  }

  // ---- gates + cumulative log-forget
  gates_kernel<<<dim3(Sc, NHc, Bc), 256, 0, stream>>>(qbf, kbf, vbf, Wi, bi,
                                                      Wf, bf, ipre, fpre);
  cumsum_kernel<<<1, 32, 0, stream>>>(fpre, lfc);

  // ---- stabilized mLSTM attention (writes h into xm)
  attn_kernel<<<dim3(Sc / 16, NHc, Bc), 256, 0, stream>>>(qbf, kbf, vtbf,
                                                          ipre, lfc, xm);

  // ---- layernorm + skip + swish(z) gate -> bf16 h_state
  ln_gate_kernel<<<dim3(Sc, NHc, Bc), 256, 0, stream>>>(xm, actb, zb, nsc,
                                                        skp, hstbf);

  // ---- down projection (4096 x 2048) x (2048 x 1024)
  gemm_bf16_wmma<<<dim3(Ec / 128, M / 128), 256, 0, stream>>>(
      hstbf, Wdt, out, M, Ec, INNERc);
}
